// ContTimeLSTM_58763742544916
// MI455X (gfx1250) — matrix operand order repaired and checked by reading, not compiled, not run on hardware
//
#include <hip/hip_runtime.h>
#include <hip/hip_bf16.h>

typedef __attribute__((ext_vector_type(16))) __bf16 v16bf;
typedef __attribute__((ext_vector_type(8)))  __bf16 v8bf;
typedef __attribute__((ext_vector_type(8)))  float  v8f;

#define Bsz   64
#define Lsz   1024
#define Dsz   256
#define Hsz   256
#define Ksz   512           // D + H
#define NBLK  9             // 8 gate blocks + 1 delta block
#define HSL   32            // H columns per gate block
#define NC    (6 * HSL)     // 192 GEMM columns per gate block
#define WBYTES  (NC * Ksz * 2)      // 196608 B bf16 weights (column-major)
#define XSBYTES (Bsz * Ksz * 2)     // 65536 B bf16 activations
#define SMEM_TOTAL (WBYTES + XSBYTES + 1024)

__device__ __forceinline__ float sigm(float v) { return 1.0f / (1.0f + __expf(-v)); }

// Monotonic-counter grid barrier (counter zeroed by hipMemsetAsync each call).
__device__ __forceinline__ void grid_barrier(unsigned* bar, unsigned target) {
    __threadfence();
    __syncthreads();
    if (threadIdx.x == 0) {
        __hip_atomic_fetch_add(bar, 1u, __ATOMIC_RELEASE, __HIP_MEMORY_SCOPE_AGENT);
        while (__hip_atomic_load(bar, __ATOMIC_ACQUIRE, __HIP_MEMORY_SCOPE_AGENT) < target)
            __builtin_amdgcn_s_sleep(2);
    }
    __syncthreads();
    __threadfence();
}

__global__ __launch_bounds__(256) void ctlstm_persistent(
    const float* __restrict__ x,        // (B, L, D)
    const float* __restrict__ td,       // (B, L)
    const float* W0, const float* W1, const float* W2,
    const float* W3, const float* W4, const float* W5,   // (512, 256) each
    const float* b0, const float* b1, const float* b2,
    const float* b3, const float* b4, const float* b5,   // (256,) each
    const float* Wdh, const float* bdh, const float* betap,
    float* __restrict__ out,            // (B,L,H) then (B, 3H+1)
    unsigned char* __restrict__ ws)
{
    const int tid = threadIdx.x;
    const int blk = blockIdx.x;
    const bool isDelta = (blk == 8);

    unsigned* bar  = (unsigned*)ws;
    float*    dbuf = (float*)(ws + 256);        // dbuf[phase*64 + b], zeroed each call

    extern __shared__ char smem[];
    __bf16* Wl  = (__bf16*)smem;                       // [NC][Ksz], n-major, k contiguous
    __bf16* XS  = (__bf16*)(smem + WBYTES);            // [Bsz][Ksz] row-major
    float*  sDT = (float*)(smem + WBYTES + XSBYTES);   // 64 time-deltas
    float*  sDP = sDT + 64;                            // 64 previous d values
    float*  Wd  = (float*)Wl;                          // delta block only: 512 fp32

    const float* Wg[6] = {W0, W1, W2, W3, W4, W5};
    const float* bg[6] = {b0, b1, b2, b3, b4, b5};
    const int h0 = blk * HSL;

    // ---------- one-time weight staging (fp32 -> bf16 LDS, column-major) ----------
    if (!isDelta) {
        for (int idx = tid; idx < NC * Ksz; idx += 256) {
            int n = idx >> 9;            // / Ksz
            int k = idx & (Ksz - 1);
            int gg  = n >> 5;            // gate
            int col = h0 + (n & (HSL - 1));
            Wl[idx] = (__bf16)(Wg[gg][k * Hsz + col]);
        }
    } else {
        for (int k = tid; k < Ksz; k += 256) Wd[k] = Wdh[k];   // W_d is (512,1)
    }
    __syncthreads();

    // ---------- wave / lane mapping ----------
    const int wv   = tid >> 5;
    const int ln   = tid & 31;
    const int mrow = wv & 3;          // row tile: rows [16m, 16m+16)
    const int g2   = wv >> 2;         // h-subgroup: cols [16g2, 16g2+16) of the 32-col slice
    const int colB = 16 * g2;
    const int nA   = ln & 15;
    const int hiL  = ln >> 4;
    const int rbase = 16 * mrow + 8 * hiL;     // C-layout: reg r -> batch row rbase + r
    const int jcol  = h0 + colB + nA;          // global H column owned by this lane

    float bsl[6];                               // bias per gate for this lane's column
    if (!isDelta) {
        #pragma unroll
        for (int g = 0; g < 6; ++g) bsl[g] = bg[g][jcol];
    }
    const float betaV = betap[0];
    const float bdv   = bdh[0];

    // persistent recurrent state (registers for all 1024 steps)
    float cs[8], ce[8], og[8], cc[8];
    #pragma unroll
    for (int r = 0; r < 8; ++r) { cs[r] = 0.f; ce[r] = 0.f; og[r] = 0.f; cc[r] = 0.f; }
    float myD = 0.f;

    unsigned barTarget = 0;

    for (int t = 0; t < Lsz; ++t) {
        // ================= phase 1: decay + h, publish h into out[:, t, :] ==========
        if (!isDelta) {
            if (tid < Bsz) {
                sDT[tid] = td[(size_t)tid * Lsz + t];
                sDP[tid] = dbuf[((t & 1) ^ 1) * 64 + tid];
            }
            __syncthreads();
            #pragma unroll
            for (int r = 0; r < 8; ++r) {
                int b = rbase + r;
                float c = cs[r] + (ce[r] - cs[r]) * __expf(-sDP[b] * sDT[b]);
                cc[r] = c;
                float h = og[r] * tanhf(c);     // 2*sigmoid(2c)-1 == tanh(c)
                out[(size_t)b * Lsz * Hsz + (size_t)t * Hsz + jcol] = h;
            }
        }
        barTarget += NBLK;
        grid_barrier(bar, barTarget);

        // ================= phase 2: build xs = [x_t, h_t] in LDS as bf16 ============
        for (int idx = tid; idx < (Bsz * Ksz) / 4; idx += 256) {
            int b  = idx >> 7;           // / 128
            int kq = idx & 127;
            float4 v;
            if (kq < 64)
                v = ((const float4*)(x + ((size_t)b * Lsz + t) * Dsz))[kq];
            else
                v = ((const float4*)(out + (size_t)b * Lsz * Hsz + (size_t)t * Hsz))[kq - 64];
            __bf16* row = XS + (size_t)b * Ksz + kq * 4;
            row[0] = (__bf16)v.x; row[1] = (__bf16)v.y;
            row[2] = (__bf16)v.z; row[3] = (__bf16)v.w;
        }
        if (t + 1 < Lsz)  // pull next step's x_t toward the caches (global_prefetch_b8)
            __builtin_prefetch((const void*)(x + ((size_t)(tid & 63) * Lsz + (t + 1)) * Dsz
                                             + ((tid >> 6) << 6)), 0, 1);
        __syncthreads();

        if (!isDelta) {
            // ============= phase 3: bf16 WMMA GEMM, 16 k-steps of 32 ================
            v8f acc[6];
            #pragma unroll
            for (int g = 0; g < 6; ++g) { v8f z = {}; acc[g] = z; }

            const int aRow = 16 * mrow + (ln & 15);
            const __bf16* aPtr = XS + (size_t)aRow * Ksz;
            const int aOff = hiL * 16;
            const int bOff = hiL * 16;

            #pragma unroll 4
            for (int ks = 0; ks < 16; ++ks) {
                const int kb = ks * 32;
                v8bf alo = *(const v8bf*)(aPtr + kb + aOff);        // K = kb+aOff .. +7
                v8bf ahi = *(const v8bf*)(aPtr + kb + 8 + aOff);    // K = kb+8+aOff .. +7
                v16bf a = __builtin_shufflevector(alo, ahi,
                          0, 1, 2, 3, 4, 5, 6, 7, 8, 9, 10, 11, 12, 13, 14, 15);
                #pragma unroll
                for (int g = 0; g < 6; ++g) {
                    int n = g * HSL + colB + nA;
                    v16bf bf = *(const v16bf*)(Wl + (size_t)n * Ksz + kb + bOff);
                    acc[g] = __builtin_amdgcn_wmma_f32_16x16x32_bf16(
                                 false, a, false, bf, (short)0, acc[g], false, false);
                }
            }

            // ============= phase 4: gates + lane-local state update =================
            #pragma unroll
            for (int r = 0; r < 8; ++r) {
                float gi  = sigm(acc[0][r] + bsl[0]);
                float gf  = sigm(acc[1][r] + bsl[1]);
                float gie = sigm(acc[2][r] + bsl[2]);
                float gfe = sigm(acc[3][r] + bsl[3]);
                float z   = 2.0f * sigm(acc[4][r] + bsl[4]) - 1.0f;
                float go  = sigm(acc[5][r] + bsl[5]);
                cs[r] = gf * cc[r] + gi * z;
                ce[r] = gfe * ce[r] + gie * z;
                og[r] = go;
            }
        } else {
            // delta head: 64 rows, one thread per row, 512-dot against W_d
            if (tid < Bsz) {
                float s = bdv;
                const __bf16* xr = XS + (size_t)tid * Ksz;
                #pragma unroll 8
                for (int k = 0; k < Ksz; ++k) s += (float)xr[k] * Wd[k];
                float bs = betaV * s;
                float sp = (bs > 20.0f) ? bs : log1pf(__expf(bs));
                myD = sp / betaV;
                dbuf[(t & 1) * 64 + tid] = myD;
            }
        }
        barTarget += NBLK;
        grid_barrier(bar, barTarget);
    }

    // ---------- final states: (1, B, 3H+1) appended after outputs ----------
    const size_t so = (size_t)Bsz * Lsz * Hsz;
    if (!isDelta) {
        #pragma unroll
        for (int r = 0; r < 8; ++r) {
            size_t base = so + (size_t)(rbase + r) * (3 * Hsz + 1);
            out[base + jcol]           = og[r];
            out[base + Hsz + jcol]     = cs[r];
            out[base + 2 * Hsz + jcol] = ce[r];
        }
    } else if (tid < Bsz) {
        out[so + (size_t)tid * (3 * Hsz + 1) + 3 * Hsz] = myD;
    }
}

extern "C" void kernel_launch(void* const* d_in, const int* in_sizes, int n_in,
                              void* d_out, int out_size, void* d_ws, size_t ws_size,
                              hipStream_t stream) {
    (void)in_sizes; (void)n_in; (void)out_size; (void)ws_size;
    const float* x    = (const float*)d_in[0];
    const float* td   = (const float*)d_in[1];
    const float* W_i  = (const float*)d_in[2];
    const float* b_i  = (const float*)d_in[3];
    const float* W_f  = (const float*)d_in[4];
    const float* b_f  = (const float*)d_in[5];
    const float* W_ie = (const float*)d_in[6];
    const float* b_ie = (const float*)d_in[7];
    const float* W_fe = (const float*)d_in[8];
    const float* b_fe = (const float*)d_in[9];
    const float* W_z  = (const float*)d_in[10];
    const float* b_z  = (const float*)d_in[11];
    const float* W_o  = (const float*)d_in[12];
    const float* b_o  = (const float*)d_in[13];
    const float* W_d  = (const float*)d_in[14];
    const float* b_d  = (const float*)d_in[15];
    const float* beta = (const float*)d_in[16];

    // Deterministically zero the barrier counter + d double-buffer (capture-legal).
    hipMemsetAsync(d_ws, 0, 1024, stream);

    ctlstm_persistent<<<dim3(NBLK), dim3(256), SMEM_TOTAL, stream>>>(
        x, td, W_i, W_f, W_ie, W_fe, W_z, W_o,
        b_i, b_f, b_ie, b_fe, b_z, b_o,
        W_d, b_d, beta,
        (float*)d_out, (unsigned char*)d_ws);
}
